// RadialProfileModel_86234353369344
// MI455X (gfx1250) — matrix operand order repaired and checked by reading, not compiled, not run on hardware
//
#include <hip/hip_runtime.h>
#include <hip/hip_bf16.h>
#include <math.h>

// ---------------------------------------------------------------------------
// Types for CDNA5 WMMA (wave32)
// ---------------------------------------------------------------------------
typedef __attribute__((ext_vector_type(16))) __bf16 v16bf;
typedef __attribute__((ext_vector_type(8)))  __bf16 v8bf;
typedef __attribute__((ext_vector_type(4)))  __bf16 v4bf;
typedef __attribute__((ext_vector_type(8)))  float  v8f;

static __device__ __forceinline__ v8bf ld8(const __bf16* p) {
    return *(const v8bf*)p;   // 16B aligned by construction (global or LDS)
}
static __device__ __forceinline__ v16bf cat8(v8bf lo, v8bf hi) {
    return __builtin_shufflevector(lo, hi, 0,1,2,3,4,5,6,7,8,9,10,11,12,13,14,15);
}
static __device__ __forceinline__ v8f wmma_bf16(v16bf a, v16bf b, v8f c) {
    // emits v_wmma_f32_16x16x32_bf16
    return __builtin_amdgcn_wmma_f32_16x16x32_bf16(false, a, false, b, (short)0, c, false, false);
}

// ---------------------------------------------------------------------------
// Problem constants
// ---------------------------------------------------------------------------
#define NN       512
#define BATCH    64
#define CHUNK    8            // images per stage1/stage2 pass
#define NBIN     384          // padded ring-bin count (363 used)
#define MAXR     256
#define LDSW     520          // padded LDS row stride in bf16 (1040B: 4-bank skew/row)

// workspace layout (bytes)
#define OFF_G     ((size_t)0)                                   // 64*512*512 bf16
#define OFF_C     (OFF_G + (size_t)BATCH*NN*NN*2)               // 512*512 bf16
#define OFF_S     (OFF_C + (size_t)NN*NN*2)                     // 512*512 bf16
#define OFF_R     (OFF_S + (size_t)NN*NN*2)                     // CHUNK*512*512 bf16
#define OFF_P     (OFF_R + (size_t)CHUNK*NN*NN*2)               // CHUNK*512*512 bf16
#define OFF_RING  (OFF_P + (size_t)CHUNK*NN*NN*2)               // 64*NBIN f32
#define OFF_CNT   (OFF_RING + (size_t)BATCH*NBIN*4)             // NBIN f32

// ---------------------------------------------------------------------------
// Tensor Data Mover: async-load a 512x16 bf16 tile (16 rows of a 512x512
// matrix) into LDS with 16B padding every 1024B row (-> LDSW stride).
// Descriptor per CDNA5 ISA sec 8.3/8.4.  This toolchain exposes the 6-arg
// builtin: (u32x4 g0, i32x8 g1, i32x4 g2, i32x4 g3, i32x8 extra, i32 cpol).
// ---------------------------------------------------------------------------
#define USE_TDM 1
#if USE_TDM
typedef __attribute__((ext_vector_type(4))) unsigned int u32x4;
typedef __attribute__((ext_vector_type(8))) int          i32x8;
typedef __attribute__((ext_vector_type(4))) int          i32x4;

static __device__ __forceinline__ void tdm_load_tile16(const __bf16* src, void* ldsDst) {
    unsigned long long ga = (unsigned long long)(const void*)src;
    unsigned lds = (unsigned)(unsigned long long)ldsDst;   // low 32 bits = LDS byte addr
    u32x4 g0 = { 1u,                                       // count=1, is_restore=0, gather=0
                 lds,                                      // lds_addr
                 (unsigned)(ga & 0xFFFFFFFFu),             // global_addr[31:0]
                 (unsigned)((ga >> 32) & 0x1FFFFFFu) | (2u << 30) };  // addr[56:32], type=2
    i32x8 g1 = { (int)((1u << 16)          // data_size = 2B
                     | (1u << 20)          // pad_enable
                     | (7u << 22)          // pad_interval: 256 DWORDs (one 1024B row)
                     | (3u << 25)),        // pad_amount: 4 DWORDs (16B)
                 (int)(512u << 16),        // tensor_dim0 = 512  (bits[79:48])
                 (int)(512u << 16),        // tensor_dim0 hi=0 | tensor_dim1 = 512 lo16
                 (int)(512u << 16),        // tensor_dim1 hi=0 | tile_dim0 = 512
                 16,                       // tile_dim1 = 16, tile_dim2 = 0
                 512,                      // tensor_dim0_stride lo32 = 512
                 0,                        // stride0 hi16 | stride1 lo16
                 4 };                      // tensor_dim1_stride[47:16] (512*512 >> 16)
    i32x4 z4 = { 0, 0, 0, 0 };             // 2-D tensor: groups 2/3 unused
    i32x8 z8 = { 0, 0, 0, 0, 0, 0, 0, 0 };
    __builtin_amdgcn_tensor_load_to_lds(g0, g1, z4, z4, z8, 0);
}
#endif

// Stage A-tiles (C and S rows) into LDS; wave0 issues the DMA (scalar branch:
// TDM ignores EXEC, so guard must be wave-granular), all waves sync after.
static __device__ __forceinline__ void stage_A_tiles(const __bf16* Cm, const __bf16* Sm,
                                                     int row0, __bf16* sC, __bf16* sS) {
#if USE_TDM
    if (__builtin_amdgcn_readfirstlane(threadIdx.x) < 32) {
        tdm_load_tile16(Cm + (size_t)row0 * NN, sC);
        tdm_load_tile16(Sm + (size_t)row0 * NN, sS);
        __builtin_amdgcn_s_wait_tensorcnt(0);
    }
    __syncthreads();
#else
    // cooperative copy fallback: 16 rows x 512 bf16, 16B chunks
    for (int idx = threadIdx.x; idx < 16 * 64; idx += 256) {
        int r = idx >> 6, c = (idx & 63) * 8;
        *(v8bf*)(sC + r * LDSW + c) = ld8(Cm + (size_t)(row0 + r) * NN + c);
        *(v8bf*)(sS + r * LDSW + c) = ld8(Sm + (size_t)(row0 + r) * NN + c);
    }
    __syncthreads();
#endif
}

// ---------------------------------------------------------------------------
// Zero ring sums + counts each launch (ws is poisoned / reused across replays)
// ---------------------------------------------------------------------------
__global__ void zero_kernel(float* ring, float* cnt) {
    int idx = blockIdx.x * 256 + threadIdx.x;
    int tot = BATCH * NBIN;
    if (idx < tot) ring[idx] = 0.0f;
    if (idx < NBIN) cnt[idx] = 0.0f;
}

// ---------------------------------------------------------------------------
// DFT cos/sin matrices (symmetric): C[i][j]=cos(2*pi*i*j/N), S=sin
// ---------------------------------------------------------------------------
__global__ void trig_kernel(__bf16* Cm, __bf16* Sm) {
    int idx = blockIdx.x * 256 + threadIdx.x;   // 512*512 threads
    int i = idx >> 9, j = idx & (NN - 1);
    int t = (i * j) & (NN - 1);                 // exploit periodicity for accuracy
    float ang = (float)t * 0.012271846303085129f;  // 2*pi/512
    Cm[idx] = (__bf16)__cosf(ang);
    Sm[idx] = (__bf16)__sinf(ang);
}

// ---------------------------------------------------------------------------
// Grayscale: x (64,3,512,512) f32 -> G (64,512,512) bf16   (memory-bound pass)
// ---------------------------------------------------------------------------
__global__ void gray_kernel(const float* __restrict__ x, __bf16* __restrict__ G) {
    size_t tid = (size_t)blockIdx.x * 256 + threadIdx.x;     // each does 4 px
    size_t e   = tid * 4;
    size_t img = e / ((size_t)NN * NN);
    size_t off = e % ((size_t)NN * NN);
    const float* base = x + img * 3 * (size_t)NN * NN + off;
    float4 r = *(const float4*)(base);
    float4 g = *(const float4*)(base + (size_t)NN * NN);
    float4 b = *(const float4*)(base + 2 * (size_t)NN * NN);
    v4bf o;
    o[0] = (__bf16)(0.2989f * r.x + 0.587f * g.x + 0.114f * b.x);
    o[1] = (__bf16)(0.2989f * r.y + 0.587f * g.y + 0.114f * b.y);
    o[2] = (__bf16)(0.2989f * r.z + 0.587f * g.z + 0.114f * b.z);
    o[3] = (__bf16)(0.2989f * r.w + 0.587f * g.w + 0.114f * b.w);
    *(v4bf*)(G + img * (size_t)NN * NN + off) = o;
}

// ---------------------------------------------------------------------------
// Ring pixel counts (computed once per launch; image independent)
// ---------------------------------------------------------------------------
__global__ void counts_kernel(float* __restrict__ cnt) {
    __shared__ float bins[NBIN];
    int t = threadIdx.x;
    for (int i = t; i < NBIN; i += 256) bins[i] = 0.0f;
    __syncthreads();
    int idx = blockIdx.x * 256 + t;             // 512*512 threads
    int v = idx >> 9, u = idx & (NN - 1);
    int fx = (u < NN / 2) ? u : u - NN;
    int fy = (v < NN / 2) ? v : v - NN;
    int rad = (int)sqrtf((float)(fx * fx + fy * fy));
    atomicAdd(&bins[rad], 1.0f);
    __syncthreads();
    for (int i = t; i < NBIN; i += 256)
        if (bins[i] != 0.0f) atomicAdd(&cnt[i], bins[i]);
}

// ---------------------------------------------------------------------------
// Stage 1:  R = C * G^T ,  P = S * G^T   (row DFT, transposed output)
// WG fixes the u-tile: A (C/S rows u0..u0+15) staged once in LDS via TDM and
// shared by all 8 waves; B (rows of G, contiguous K) streamed from global.
// ---------------------------------------------------------------------------
__global__ void stage1_kernel(const __bf16* __restrict__ G,
                              const __bf16* __restrict__ Cm,
                              const __bf16* __restrict__ Sm,
                              __bf16* __restrict__ R, __bf16* __restrict__ P,
                              int imgBase) {
    __shared__ __bf16 sC[16 * LDSW];
    __shared__ __bf16 sS[16 * LDSW];

    const int lane = threadIdx.x & 31;
    const int w    = threadIdx.x >> 5;
    const int l16  = lane & 15;
    const int hl   = lane >> 4;
    const int wg   = blockIdx.x;
    const int img  = wg >> 7;            // / 128
    const int rem  = wg & 127;
    const int u0   = (rem >> 2) * 16;
    const int y0   = ((rem & 3) * 8 + w) * 16;

    stage_A_tiles(Cm, Sm, u0, sC, sS);

    const __bf16* g = G + (size_t)(imgBase + img) * NN * NN;
    __bf16* rp = R + (size_t)img * NN * NN;
    __bf16* pp = P + (size_t)img * NN * NN;

    v8f accR = {}; v8f accP = {};
    for (int k0 = 0; k0 < NN; k0 += 32) {
        const __bf16* ca = sC + l16 * LDSW + k0 + hl * 8;        // A halves {0..7,16..23}/{8..15,24..31}
        const __bf16* sa = sS + l16 * LDSW + k0 + hl * 8;
        const __bf16* gb = g + (y0 + l16) * NN + k0 + hl * 8;    // B halves {0..7,16..23}/{8..15,24..31}
        v16bf aC = cat8(ld8(ca), ld8(ca + 16));
        v16bf aS = cat8(ld8(sa), ld8(sa + 16));
        v16bf bG = cat8(ld8(gb), ld8(gb + 16));
        accR = wmma_bf16(aC, bG, accR);
        accP = wmma_bf16(aS, bG, accP);
    }
#pragma unroll
    for (int rr = 0; rr < 8; ++rr) {
        int u = u0 + rr + hl * 8;        // C/D layout: M = r + 8*half, N = lane%16
        int y = y0 + l16;
        rp[u * NN + y] = (__bf16)accR[rr];
        pp[u * NN + y] = (__bf16)accP[rr];
    }
}

// ---------------------------------------------------------------------------
// Stage 2: column DFT + |.| + fftshift radius binning (fused epilogue).
//   re = C.R_B - S.P_B ;  im (up to sign) = C.P_B + S.R_B
// A (C/S rows v0..v0+15) staged in LDS via TDM, shared across the WG.
// ---------------------------------------------------------------------------
__global__ void stage2_kernel(const __bf16* __restrict__ R,
                              const __bf16* __restrict__ P,
                              const __bf16* __restrict__ Cm,
                              const __bf16* __restrict__ Sm,
                              float* __restrict__ ring, int imgBase) {
    __shared__ __bf16 sC[16 * LDSW];
    __shared__ __bf16 sS[16 * LDSW];
    __shared__ float bins[NBIN];

    const int t = threadIdx.x;
    for (int i = t; i < NBIN; i += 256) bins[i] = 0.0f;

    const int lane = t & 31;
    const int w    = t >> 5;
    const int l16  = lane & 15;
    const int hl   = lane >> 4;
    const int wg   = blockIdx.x;
    const int img  = wg >> 7;
    const int rem  = wg & 127;
    const int v0   = (rem >> 2) * 16;
    const int u0   = ((rem & 3) * 8 + w) * 16;

    stage_A_tiles(Cm, Sm, v0, sC, sS);   // includes the barrier (covers bins init)

    const __bf16* rp = R + (size_t)img * NN * NN;
    const __bf16* pp = P + (size_t)img * NN * NN;

    v8f a1 = {}, a2 = {}, a3 = {}, a4 = {};
    for (int k0 = 0; k0 < NN; k0 += 32) {
        const __bf16* ca = sC + l16 * LDSW + k0 + hl * 8;
        const __bf16* sa = sS + l16 * LDSW + k0 + hl * 8;
        const __bf16* rb = rp + (u0 + l16) * NN + k0 + hl * 8;   // B from R (already transposed)
        const __bf16* pb = pp + (u0 + l16) * NN + k0 + hl * 8;   // B from P
        v16bf aC = cat8(ld8(ca), ld8(ca + 16));
        v16bf aS = cat8(ld8(sa), ld8(sa + 16));
        v16bf bR = cat8(ld8(rb), ld8(rb + 16));
        v16bf bP = cat8(ld8(pb), ld8(pb + 16));
        a1 = wmma_bf16(aC, bR, a1);
        a2 = wmma_bf16(aS, bP, a2);
        a3 = wmma_bf16(aC, bP, a3);
        a4 = wmma_bf16(aS, bR, a4);
    }
    const float inv = 1.0f / ((float)NN * (float)NN);   // fft norm='forward'
#pragma unroll
    for (int rr = 0; rr < 8; ++rr) {
        float re  = a1[rr] - a2[rr];
        float im  = a3[rr] + a4[rr];                    // sign irrelevant for |.|
        float mag = sqrtf(re * re + im * im) * inv;
        int v = v0 + rr + hl * 8;
        int u = u0 + l16;
        int fx = (u < NN / 2) ? u : u - NN;             // fftshift radius
        int fy = (v < NN / 2) ? v : v - NN;
        int rad = (int)sqrtf((float)(fx * fx + fy * fy));
        atomicAdd(&bins[rad], mag);
    }
    __syncthreads();
    for (int i = t; i < NBIN; i += 256)
        if (bins[i] != 0.0f) atomicAdd(&ring[(size_t)(imgBase + img) * NBIN + i], bins[i]);
}

// ---------------------------------------------------------------------------
// Head: ring means -> log1p -> min-max norm -> conv1d x3 (+relu,+pool2) ->
// GAP -> FC.  One 256-thread block per image (trivial FLOPs).
// ---------------------------------------------------------------------------
__global__ void head_kernel(const float* __restrict__ ring, const float* __restrict__ cnt,
                            const float* __restrict__ w1, const float* __restrict__ b1,
                            const float* __restrict__ w2, const float* __restrict__ b2,
                            const float* __restrict__ w3, const float* __restrict__ b3,
                            const float* __restrict__ fw, const float* __restrict__ fb,
                            float* __restrict__ out) {
    const int b = blockIdx.x;
    const int t = threadIdx.x;                  // 256 threads
    __shared__ float pn[MAXR + 2];
    __shared__ float sA[4096];
    __shared__ float sH[2048];
    __shared__ float red[256];

    float s = ring[(size_t)b * NBIN + t];
    float c = cnt[t];
    float lg = log1pf(c > 0.0f ? s / c : 0.0f);

    red[t] = lg; __syncthreads();
    for (int o = 128; o > 0; o >>= 1) { if (t < o) red[t] = fminf(red[t], red[t + o]); __syncthreads(); }
    float mn = red[0]; __syncthreads();
    red[t] = lg; __syncthreads();
    for (int o = 128; o > 0; o >>= 1) { if (t < o) red[t] = fmaxf(red[t], red[t + o]); __syncthreads(); }
    float mx = red[0]; __syncthreads();
    float rng = mx - mn;
    pn[t + 1] = (rng > 0.0f) ? (lg - mn) / rng : 0.0f;
    if (t == 0) { pn[0] = 0.0f; pn[MAXR + 1] = 0.0f; }
    __syncthreads();

    // conv1: (16,1,3), pad 1, relu
    for (int idx = t; idx < 16 * 256; idx += 256) {
        int ch = idx >> 8, i = idx & 255;
        float a = b1[ch] + w1[ch*3+0]*pn[i] + w1[ch*3+1]*pn[i+1] + w1[ch*3+2]*pn[i+2];
        sA[idx] = fmaxf(a, 0.0f);
    }
    __syncthreads();
    for (int idx = t; idx < 16 * 128; idx += 256) {        // maxpool2 -> 16x128
        int ch = idx >> 7, i = idx & 127;
        sH[idx] = fmaxf(sA[ch*256 + 2*i], sA[ch*256 + 2*i + 1]);
    }
    __syncthreads();
    // conv2: (32,16,3), pad 1, relu
    for (int idx = t; idx < 32 * 128; idx += 256) {
        int ch = idx >> 7, i = idx & 127;
        float a = b2[ch];
        for (int ic = 0; ic < 16; ++ic) {
            const float* wv = w2 + (ch*16 + ic) * 3;
            float xm = (i > 0)   ? sH[ic*128 + i - 1] : 0.0f;
            float x0 =             sH[ic*128 + i];
            float xp = (i < 127) ? sH[ic*128 + i + 1] : 0.0f;
            a += wv[0]*xm + wv[1]*x0 + wv[2]*xp;
        }
        sA[idx] = fmaxf(a, 0.0f);
    }
    __syncthreads();
    for (int idx = t; idx < 32 * 64; idx += 256) {         // maxpool2 -> 32x64
        int ch = idx >> 6, i = idx & 63;
        sH[idx] = fmaxf(sA[ch*128 + 2*i], sA[ch*128 + 2*i + 1]);
    }
    __syncthreads();
    // conv3: (64,32,3), pad 1, relu
    for (int idx = t; idx < 64 * 64; idx += 256) {
        int ch = idx >> 6, i = idx & 63;
        float a = b3[ch];
        for (int ic = 0; ic < 32; ++ic) {
            const float* wv = w3 + (ch*32 + ic) * 3;
            float xm = (i > 0)  ? sH[ic*64 + i - 1] : 0.0f;
            float x0 =            sH[ic*64 + i];
            float xp = (i < 63) ? sH[ic*64 + i + 1] : 0.0f;
            a += wv[0]*xm + wv[1]*x0 + wv[2]*xp;
        }
        sA[idx] = fmaxf(a, 0.0f);
    }
    __syncthreads();
    // GAP + FC
    if (t < 64) {
        float g = 0.0f;
        for (int i = 0; i < 64; ++i) g += sA[t*64 + i];
        red[t] = (g * (1.0f / 64.0f)) * fw[t];
    }
    __syncthreads();
    for (int o = 32; o > 0; o >>= 1) { if (t < o && t + o < 64) red[t] += red[t + o]; __syncthreads(); }
    if (t == 0) out[b] = red[0] + fb[0];
}

// ---------------------------------------------------------------------------
// Launch
// ---------------------------------------------------------------------------
extern "C" void kernel_launch(void* const* d_in, const int* in_sizes, int n_in,
                              void* d_out, int out_size, void* d_ws, size_t ws_size,
                              hipStream_t stream) {
    (void)in_sizes; (void)n_in; (void)out_size; (void)ws_size;
    const float* x  = (const float*)d_in[0];
    const float* w1 = (const float*)d_in[1];
    const float* b1 = (const float*)d_in[2];
    const float* w2 = (const float*)d_in[3];
    const float* b2 = (const float*)d_in[4];
    const float* w3 = (const float*)d_in[5];
    const float* b3 = (const float*)d_in[6];
    const float* fw = (const float*)d_in[7];
    const float* fb = (const float*)d_in[8];
    float* out = (float*)d_out;

    char* ws = (char*)d_ws;
    __bf16* G    = (__bf16*)(ws + OFF_G);
    __bf16* Cm   = (__bf16*)(ws + OFF_C);
    __bf16* Sm   = (__bf16*)(ws + OFF_S);
    __bf16* R    = (__bf16*)(ws + OFF_R);
    __bf16* P    = (__bf16*)(ws + OFF_P);
    float*  ring = (float*)(ws + OFF_RING);
    float*  cnt  = (float*)(ws + OFF_CNT);

    zero_kernel<<<(BATCH * NBIN + 255) / 256, 256, 0, stream>>>(ring, cnt);
    trig_kernel<<<(NN * NN) / 256, 256, 0, stream>>>(Cm, Sm);
    gray_kernel<<<(BATCH * NN * NN / 4) / 256, 256, 0, stream>>>(x, G);
    counts_kernel<<<(NN * NN) / 256, 256, 0, stream>>>(cnt);

    for (int c0 = 0; c0 < BATCH; c0 += CHUNK) {
        stage1_kernel<<<CHUNK * 128, 256, 0, stream>>>(G, Cm, Sm, R, P, c0);
        stage2_kernel<<<CHUNK * 128, 256, 0, stream>>>(R, P, Cm, Sm, ring, c0);
    }
    head_kernel<<<BATCH, 256, 0, stream>>>(ring, cnt, w1, b1, w2, b2, w3, b3, fw, fb, out);
}
